// TQMamba_40097814675765
// MI455X (gfx1250) — matrix operand order repaired and checked by reading, not compiled, not run on hardware
//
#include <hip/hip_runtime.h>
#include <hip/hip_bf16.h>
#include <math.h>
#include <stdint.h>

// ---------------- model constants ----------------
#define NL 12
#define DM 768
#define DI 1536
#define DS 16
#define DC 4
#define DR 48
#define BB 4
#define LSEQ 2048
#define TOK (BB * LSEQ)        // 8192 rows
#define NXZ (2 * DI)           // 3072
#define NXP 96                 // x_proj N padded 80 -> 96
#define KDT 64                 // dt_proj K padded 48 -> 64
#define LN_EPS 1e-6f
#define SCAN_CH 128            // scan L-chunk staged in LDS via TDM

typedef __attribute__((ext_vector_type(16))) __bf16 v16bf;
typedef __attribute__((ext_vector_type(8)))  __bf16 v8bf;
typedef __attribute__((ext_vector_type(8)))  float  v8f;
typedef __attribute__((ext_vector_type(4)))  unsigned int v4u;
typedef __attribute__((ext_vector_type(8)))  int v8i;
typedef __attribute__((ext_vector_type(4)))  int v4i;

// ---------------- WMMA tile loaders (CDNA5 bf16 16x16x32 layouts) ----------------
// A-matrix 16x32 bf16: lane m (0-15): V0-3 = K0..7, V4-7 = K16..23
//                      lane m+16    : V0-3 = K8..15, V4-7 = K24..31
__device__ __forceinline__ v16bf load_a_tile(const __bf16* __restrict__ A, int lda,
                                             int row0, int k0, int lane) {
  int m    = row0 + (lane & 15);
  int half = lane >> 4;
  const __bf16* p = A + (size_t)m * lda + k0 + 8 * half;
  v8bf lo = *(const v8bf*)p;          // K = k0+8h .. k0+8h+7
  v8bf hi = *(const v8bf*)(p + 16);   // K = k0+16+8h .. +7
  v16bf r;
#pragma unroll
  for (int i = 0; i < 8; ++i) { r[i] = lo[i]; r[i + 8] = hi[i]; }
  return r;
}

// B-matrix 32x16 bf16 (B[k,n] = W[n,k], W row-major [N,K]):
// lane n (0-15): V0-3 = K0..7, V4-7 = K16..23 ; lane n+16: V0-3 = K8..15, V4-7 = K24..31
__device__ __forceinline__ v16bf load_b_tile(const __bf16* __restrict__ W, int ldw,
                                             int col0, int k0, int lane) {
  int n    = col0 + (lane & 15);
  int koff = k0 + 8 * (lane >> 4);
  const __bf16* p = W + (size_t)n * ldw + koff;
  v8bf lo = *(const v8bf*)p;
  v8bf hi = *(const v8bf*)(p + 16);
  v16bf r;
#pragma unroll
  for (int i = 0; i < 8; ++i) { r[i] = lo[i]; r[i + 8] = hi[i]; }
  return r;
}

// C/D 16x16 f32: lane n=i&15; VGPR r -> row m = r + 8*(i>>4)
__device__ __forceinline__ void store_c_tile(float* __restrict__ C, int ldc,
                                             int m0, int n0, int lane, v8f acc) {
  int col   = n0 + (lane & 15);
  int mbase = m0 + 8 * (lane >> 4);
#pragma unroll
  for (int r = 0; r < 8; ++r) C[(size_t)(mbase + r) * ldc + col] = acc[r];
}

// ---------------- GEMM 64x64/wave: C[M,N] = A[M,K] * W[N,K]^T ----------------
// 16 WMMAs per 8KB of operand traffic; grid = (N/64, M/64); blockDim = 32
__global__ __launch_bounds__(32) void gemm64_bf16_wmma(
    float* __restrict__ C, const __bf16* __restrict__ A,
    const __bf16* __restrict__ W, int M, int N, int K) {
  int lane = threadIdx.x & 31;
  int n0 = blockIdx.x * 64;
  int m0 = blockIdx.y * 64;
  v8f acc[4][4] = {};
  for (int k = 0; k < K; k += 32) {
    if (k + 64 <= K) {  // global_prefetch_b8 for next K-block (near-temporal)
      __builtin_prefetch((const void*)(A + (size_t)(m0 + (lane & 15)) * K + k + 32), 0, 3);
      __builtin_prefetch((const void*)(W + (size_t)(n0 + (lane & 15)) * K + k + 32), 0, 3);
    }
    v16bf a[4], b[4];
#pragma unroll
    for (int t = 0; t < 4; ++t) a[t] = load_a_tile(A, K, m0 + 16 * t, k, lane);
#pragma unroll
    for (int t = 0; t < 4; ++t) b[t] = load_b_tile(W, K, n0 + 16 * t, k, lane);
#pragma unroll
    for (int i = 0; i < 4; ++i)
#pragma unroll
      for (int j = 0; j < 4; ++j)
        acc[i][j] = __builtin_amdgcn_wmma_f32_16x16x32_bf16(
            false, a[i], false, b[j], (short)0, acc[i][j], false, false);
  }
#pragma unroll
  for (int i = 0; i < 4; ++i)
#pragma unroll
    for (int j = 0; j < 4; ++j)
      store_c_tile(C, N, m0 + 16 * i, n0 + 16 * j, lane, acc[i][j]);
}

// ---------------- GEMM 32x32/wave (for N not divisible by 64) ----------------
__global__ __launch_bounds__(32) void gemm32_bf16_wmma(
    float* __restrict__ C, const __bf16* __restrict__ A,
    const __bf16* __restrict__ W, int M, int N, int K) {
  int lane = threadIdx.x & 31;
  int n0 = blockIdx.x * 32;
  int m0 = blockIdx.y * 32;
  v8f acc00 = {}, acc01 = {}, acc10 = {}, acc11 = {};
  for (int k = 0; k < K; k += 32) {
    v16bf a0 = load_a_tile(A, K, m0,      k, lane);
    v16bf a1 = load_a_tile(A, K, m0 + 16, k, lane);
    v16bf b0 = load_b_tile(W, K, n0,      k, lane);
    v16bf b1 = load_b_tile(W, K, n0 + 16, k, lane);
    acc00 = __builtin_amdgcn_wmma_f32_16x16x32_bf16(false, a0, false, b0, (short)0, acc00, false, false);
    acc01 = __builtin_amdgcn_wmma_f32_16x16x32_bf16(false, a0, false, b1, (short)0, acc01, false, false);
    acc10 = __builtin_amdgcn_wmma_f32_16x16x32_bf16(false, a1, false, b0, (short)0, acc10, false, false);
    acc11 = __builtin_amdgcn_wmma_f32_16x16x32_bf16(false, a1, false, b1, (short)0, acc11, false, false);
  }
  store_c_tile(C, N, m0,      n0,      lane, acc00);
  store_c_tile(C, N, m0,      n0 + 16, lane, acc01);
  store_c_tile(C, N, m0 + 16, n0,      lane, acc10);
  store_c_tile(C, N, m0 + 16, n0 + 16, lane, acc11);
}

// ---------------- f32 -> bf16 convert with row/col zero padding ----------------
__global__ void cvt_pad_bf16(__bf16* __restrict__ dst, const float* __restrict__ src,
                             int drows, int dcols, int sstride, int crows, int ccols) {
  int idx = blockIdx.x * blockDim.x + threadIdx.x;
  int n = drows * dcols;
  if (idx >= n) return;
  int r = idx / dcols, c = idx - r * dcols;
  float v = (r < crows && c < ccols) ? src[(size_t)r * sstride + c] : 0.f;
  dst[idx] = (__bf16)v;
}

// ---------------- causal depthwise conv1d (k=4) + bias + SiLU ----------------
__global__ void conv_silu_kernel(float* __restrict__ xc, __bf16* __restrict__ xcb,
                                 const float* __restrict__ xz,
                                 const float* __restrict__ cw,
                                 const float* __restrict__ cb) {
  int idx = blockIdx.x * blockDim.x + threadIdx.x;   // over TOK*DI
  if (idx >= TOK * DI) return;
  int d  = idx % DI;
  int bl = idx / DI;
  int l  = bl % LSEQ;
  float acc = cb[d];
#pragma unroll
  for (int j = 0; j < DC; ++j) {
    int ll = l - (DC - 1) + j;
    if (ll >= 0)
      acc += xz[(size_t)(bl - (DC - 1) + j) * NXZ + d] * cw[d * DC + j];
  }
  float s = acc / (1.f + __expf(-acc));   // silu
  xc[(size_t)bl * DI + d]  = s;
  xcb[(size_t)bl * DI + d] = (__bf16)s;
}

// ---------------- dt bias + softplus (in place) ----------------
__global__ void dt_softplus_kernel(float* __restrict__ dt, const float* __restrict__ dtb) {
  int idx = blockIdx.x * blockDim.x + threadIdx.x;
  if (idx >= TOK * DI) return;
  int d = idx % DI;
  float v = dt[idx] + dtb[d];
  dt[idx] = (v > 20.f) ? v : log1pf(__expf(v));
}

// ---------------- TDM: stage a [rows x 32 f32] tile of xdbl's B|C slice into LDS ---
__device__ __forceinline__ void tdm_load_bc_chunk(unsigned int lds_off,
                                                  const float* gsrc, int rows) {
  unsigned long long ga = (unsigned long long)(uintptr_t)gsrc;
  v4u g0;
  g0[0] = 1u;                                         // count=1, user descriptor
  g0[1] = lds_off;                                    // lds_addr (bytes)
  g0[2] = (unsigned int)(ga & 0xFFFFFFFFull);         // global_addr[31:0]
  g0[3] = (unsigned int)((ga >> 32) & 0x1FFFFFFull)   // global_addr[56:32]
          | (2u << 30);                               // type = 2 ("image")
  unsigned int td0 = 32u, td1 = (unsigned int)rows;   // tensor dims (elements)
  unsigned int tl0 = 32u, tl1 = (unsigned int)rows;   // tile dims
  unsigned long long s0 = (unsigned long long)NXP;    // dim0 stride (elements)
  v8i g1;
  g1[0] = (int)(2u << 16);                            // data_size = 4B
  g1[1] = (int)((td0 & 0xFFFFu) << 16);               // tensor_dim0[15:0] @ bits 63:48
  g1[2] = (int)((td0 >> 16) | ((td1 & 0xFFFFu) << 16));
  g1[3] = (int)((td1 >> 16) | (tl0 << 16));           // tile_dim0 @ bits 127:112
  g1[4] = (int)(tl1);                                 // tile_dim1; tile_dim2=0
  g1[5] = (int)(s0 & 0xFFFFFFFFull);                  // tensor_dim0_stride[31:0]
  g1[6] = (int)((s0 >> 32) & 0xFFFFull);              // stride0[47:32]; stride1 lo = 0
  g1[7] = 0;
  v4i g2 = {0, 0, 0, 0}, g3 = {0, 0, 0, 0};
#if __clang_major__ >= 23
  v8i g4 = {0, 0, 0, 0, 0, 0, 0, 0};
  __builtin_amdgcn_tensor_load_to_lds(g0, g1, g2, g3, g4, 0);
#else
  __builtin_amdgcn_tensor_load_to_lds(g0, g1, g2, g3, 0);
#endif
}

// ---------------- selective scan, fused with D-term and silu(z) gate ----------------
// thread = one (batch, channel); B/C rows DMA'd into LDS by the TDM per L-chunk
__global__ __launch_bounds__(256) void scan_kernel(
    __bf16* __restrict__ yb, const float* __restrict__ xc,
    const float* __restrict__ dt, const float* __restrict__ xdbl,
    const float* __restrict__ xz, const float* __restrict__ A_log,
    const float* __restrict__ Dp) {
  __shared__ float bc_buf[SCAN_CH * 32];              // [chunk_row][B(16)|C(16)]
  int d = blockIdx.x * blockDim.x + threadIdx.x;      // 0..DI-1
  int b = blockIdx.y;
  float a[DS], h[DS];
#pragma unroll
  for (int s = 0; s < DS; ++s) { a[s] = -__expf(A_log[d * DS + s]); h[s] = 0.f; }
  float Dv = Dp[d];
  size_t base = (size_t)b * LSEQ;
  unsigned int lds_off = (unsigned int)(uintptr_t)&bc_buf[0];
  for (int c0 = 0; c0 < LSEQ; c0 += SCAN_CH) {
    __syncthreads();                                  // protect LDS reuse
    if (threadIdx.x < 32) {                           // wave 0 issues the DMA
      tdm_load_bc_chunk(lds_off, xdbl + (base + c0) * NXP + DR, SCAN_CH);
      __builtin_amdgcn_s_wait_tensorcnt(0);
    }
    __syncthreads();
    for (int lo = 0; lo < SCAN_CH; ++lo) {
      size_t bl = base + c0 + lo;
      float dtv = dt[bl * DI + d];
      float xv  = xc[bl * DI + d];
      float dtx = dtv * xv;
      const float* row = &bc_buf[lo * 32];            // [B(16)|C(16)]
      float y = 0.f;
#pragma unroll
      for (int s = 0; s < DS; ++s) {
        float dA = __expf(dtv * a[s]);
        h[s] = dA * h[s] + dtx * row[s];
        y += h[s] * row[16 + s];
      }
      y += xv * Dv;
      float zv  = xz[bl * NXZ + DI + d];
      float sil = zv / (1.f + __expf(-zv));
      yb[bl * DI + d] = (__bf16)(y * sil);
    }
  }
}

// ---------------- LayerNorm over DM + residual add into out ----------------
__global__ __launch_bounds__(256) void ln_add_kernel(
    float* __restrict__ out, const float* __restrict__ m,
    const float* __restrict__ w, const float* __restrict__ b) {
  __shared__ float sbuf[256];
  int row = blockIdx.x;
  int tid = threadIdx.x;
  const float* mr = m + (size_t)row * DM;
  float local = 0.f;
  for (int c = tid; c < DM; c += 256) local += mr[c];
  sbuf[tid] = local; __syncthreads();
  for (int s = 128; s > 0; s >>= 1) { if (tid < s) sbuf[tid] += sbuf[tid + s]; __syncthreads(); }
  float mu = sbuf[0] / (float)DM;
  __syncthreads();
  float lv = 0.f;
  for (int c = tid; c < DM; c += 256) { float dd = mr[c] - mu; lv += dd * dd; }
  sbuf[tid] = lv; __syncthreads();
  for (int s = 128; s > 0; s >>= 1) { if (tid < s) sbuf[tid] += sbuf[tid + s]; __syncthreads(); }
  float rstd = rsqrtf(sbuf[0] / (float)DM + LN_EPS);
  for (int c = tid; c < DM; c += 256)
    out[(size_t)row * DM + c] += (mr[c] - mu) * rstd * w[c] + b[c];
}

// ---------------- host orchestration ----------------
extern "C" void kernel_launch(void* const* d_in, const int* in_sizes, int n_in,
                              void* d_out, int out_size, void* d_ws, size_t ws_size,
                              hipStream_t stream) {
  (void)in_sizes; (void)n_in; (void)out_size; (void)ws_size;
  const float* x         = (const float*)d_in[0];
  const float* in_proj_w = (const float*)d_in[1];   // [NL, 2*DI, DM]
  const float* conv_w    = (const float*)d_in[2];   // [NL, DI, DC]
  const float* conv_b    = (const float*)d_in[3];   // [NL, DI]
  const float* x_proj_w  = (const float*)d_in[4];   // [NL, 80, DI]
  const float* dt_proj_w = (const float*)d_in[5];   // [NL, DI, DR]
  const float* dt_proj_b = (const float*)d_in[6];   // [NL, DI]
  const float* A_log     = (const float*)d_in[7];   // [NL, DI, DS]
  const float* Dp        = (const float*)d_in[8];   // [NL, DI]
  const float* out_w     = (const float*)d_in[9];   // [NL, DM, DI]
  const float* ln_w      = (const float*)d_in[10];  // [NL, DM]
  const float* ln_b      = (const float*)d_in[11];  // [NL, DM]
  float* out = (float*)d_out;                       // [TOK, DM]

  // workspace carving (256B aligned)
  char* ws = (char*)d_ws;
  size_t off = 0;
  auto carve = [&](size_t bytes) -> void* {
    void* p = ws + off;
    off = (off + bytes + 255) & ~(size_t)255;
    return p;
  };
  float*  xz   = (float*) carve((size_t)TOK * NXZ * 4);
  float*  xc   = (float*) carve((size_t)TOK * DI * 4);
  __bf16* xcb  = (__bf16*)carve((size_t)TOK * DI * 2);
  float*  xdbl = (float*) carve((size_t)TOK * NXP * 4);
  __bf16* dtrb = (__bf16*)carve((size_t)TOK * KDT * 2);
  float*  dt   = (float*) carve((size_t)TOK * DI * 4);
  __bf16* yb   = (__bf16*)carve((size_t)TOK * DI * 2);
  float*  mbuf = (float*) carve((size_t)TOK * DM * 4);
  __bf16* ab   = (__bf16*)carve((size_t)TOK * DM * 2);
  __bf16* w1b  = (__bf16*)carve((size_t)NXZ * DM * 2);
  __bf16* w2b  = (__bf16*)carve((size_t)NXP * DI * 2);
  __bf16* w3b  = (__bf16*)carve((size_t)DI * KDT * 2);
  __bf16* w4b  = (__bf16*)carve((size_t)DM * DI * 2);

  // out = x (residual stream accumulator)
  hipMemcpyAsync(out, x, (size_t)TOK * DM * 4, hipMemcpyDeviceToDevice, stream);

  const int CT = 256;
  auto nblk = [](long n, int t) { return (int)((n + t - 1) / t); };

  for (int layer = 0; layer < NL; ++layer) {
    // --- bf16 conversions / padding ---
    cvt_pad_bf16<<<nblk((long)TOK * DM, CT), CT, 0, stream>>>(
        ab, out, TOK, DM, DM, TOK, DM);
    cvt_pad_bf16<<<nblk((long)NXZ * DM, CT), CT, 0, stream>>>(
        w1b, in_proj_w + (size_t)layer * NXZ * DM, NXZ, DM, DM, NXZ, DM);
    cvt_pad_bf16<<<nblk((long)NXP * DI, CT), CT, 0, stream>>>(
        w2b, x_proj_w + (size_t)layer * (DR + 2 * DS) * DI, NXP, DI, DI, DR + 2 * DS, DI);
    cvt_pad_bf16<<<nblk((long)DI * KDT, CT), CT, 0, stream>>>(
        w3b, dt_proj_w + (size_t)layer * DI * DR, DI, KDT, DR, DI, DR);
    cvt_pad_bf16<<<nblk((long)DM * DI, CT), CT, 0, stream>>>(
        w4b, out_w + (size_t)layer * DM * DI, DM, DI, DI, DM, DI);

    // --- in_proj: xz[TOK, 3072] = ab @ w1b^T ---
    gemm64_bf16_wmma<<<dim3(NXZ / 64, TOK / 64), 32, 0, stream>>>(xz, ab, w1b, TOK, NXZ, DM);

    // --- causal depthwise conv + silu ---
    conv_silu_kernel<<<nblk((long)TOK * DI, CT), CT, 0, stream>>>(
        xc, xcb, xz, conv_w + (size_t)layer * DI * DC, conv_b + (size_t)layer * DI);

    // --- x_proj: xdbl[TOK, 96] = xcb @ w2b^T (N=96 -> 32-wide kernel) ---
    gemm32_bf16_wmma<<<dim3(NXP / 32, TOK / 32), 32, 0, stream>>>(xdbl, xcb, w2b, TOK, NXP, DI);

    // --- dt_r (cols 0..47 of xdbl) -> bf16, K padded to 64 ---
    cvt_pad_bf16<<<nblk((long)TOK * KDT, CT), CT, 0, stream>>>(
        dtrb, xdbl, TOK, KDT, NXP, TOK, DR);

    // --- dt_proj: dt[TOK, DI] = dtrb @ w3b^T; then + bias, softplus ---
    gemm64_bf16_wmma<<<dim3(DI / 64, TOK / 64), 32, 0, stream>>>(dt, dtrb, w3b, TOK, DI, KDT);
    dt_softplus_kernel<<<nblk((long)TOK * DI, CT), CT, 0, stream>>>(
        dt, dt_proj_b + (size_t)layer * DI);

    // --- selective scan (TDM-staged B/C, fused D-term + silu(z) gate) ---
    scan_kernel<<<dim3(DI / 256, BB), 256, 0, stream>>>(
        yb, xc, dt, xdbl, xz, A_log + (size_t)layer * DI * DS, Dp + (size_t)layer * DI);

    // --- out_proj: mbuf[TOK, DM] = yb @ w4b^T ---
    gemm64_bf16_wmma<<<dim3(DM / 64, TOK / 64), 32, 0, stream>>>(mbuf, yb, w4b, TOK, DM, DI);

    // --- out += LayerNorm(mbuf) ---
    ln_add_kernel<<<TOK, 256, 0, stream>>>(
        out, mbuf, ln_w + (size_t)layer * DM, ln_b + (size_t)layer * DM);
  }
}
